// mplstm_48713519071863
// MI455X (gfx1250) — matrix operand order repaired and checked by reading, not compiled, not run on hardware
//
#include <hip/hip_runtime.h>
#include <hip/hip_bf16.h>
#include <cstdint>
#include <cstddef>

#define H      1024
#define T      128
#define BB     32
#define CACHE  20
#define KROWS  (T + CACHE)   /* 148  */
#define MQ     (BB * T)      /* 4096 */
#define MK     (BB * KROWS)  /* 4736 */
#define LSTM_BLOCKS 64

typedef __attribute__((ext_vector_type(16))) _Float16 v16h;
typedef __attribute__((ext_vector_type(8)))  _Float16 v8h;
typedef __attribute__((ext_vector_type(8)))  float    v8f;
typedef __attribute__((ext_vector_type(4)))  int      v4i;

#define AS1 __attribute__((address_space(1)))
#define AS3 __attribute__((address_space(3)))

// CDNA5 async global->LDS copy path (ASYNCcnt), with safe fallback.
#if __has_builtin(__builtin_amdgcn_global_load_async_to_lds_b128) && \
    __has_builtin(__builtin_amdgcn_s_wait_asynccnt)
#define USE_ASYNC_LDS 1
#else
#define USE_ASYNC_LDS 0
#endif

__device__ __forceinline__ float sigf(float x) { return 1.0f / (1.0f + __expf(-x)); }

// ---------------------------------------------------------------------------
// f32 -> f16 conversion (vectorized by 4)
// ---------------------------------------------------------------------------
__global__ void f32_to_f16_kernel(const float* __restrict__ x,
                                  _Float16* __restrict__ y, int n4) {
  int i = blockIdx.x * 256 + threadIdx.x;
  if (i >= n4) return;
  float4 v = *(const float4*)(x + (size_t)i * 4);
  _Float16* o = y + (size_t)i * 4;
  o[0] = (_Float16)v.x; o[1] = (_Float16)v.y;
  o[2] = (_Float16)v.z; o[3] = (_Float16)v.w;
}

// ---------------------------------------------------------------------------
// LayerNorm over H=1024, one 256-thread block per row, f16 output for WMMA.
// mode 0: contiguous rows.  mode 1: row r maps to kbuf[b=r/T][CACHE + r%T].
// ---------------------------------------------------------------------------
__launch_bounds__(256)
__global__ void layernorm_kernel(const float* __restrict__ in,
                                 const float* __restrict__ gamma,
                                 const float* __restrict__ beta,
                                 _Float16* __restrict__ out, int mode) {
  const int r = blockIdx.x;
  const float* src = in + (mode ? (((size_t)(r / T) * KROWS + CACHE + (r % T)) * H)
                                : ((size_t)r * H));
  const int i0 = threadIdx.x * 4;
  float4 v = *(const float4*)(src + i0);
  float s = v.x + v.y + v.z + v.w;
  float q = v.x * v.x + v.y * v.y + v.z * v.z + v.w * v.w;
#pragma unroll
  for (int off = 16; off > 0; off >>= 1) {
    s += __shfl_xor(s, off, 32);
    q += __shfl_xor(q, off, 32);
  }
  __shared__ float ss[8], sq[8], stat[2];
  if ((threadIdx.x & 31) == 0) { ss[threadIdx.x >> 5] = s; sq[threadIdx.x >> 5] = q; }
  __syncthreads();
  if (threadIdx.x == 0) {
    float S = 0.f, Q = 0.f;
#pragma unroll
    for (int i = 0; i < 8; ++i) { S += ss[i]; Q += sq[i]; }
    float mean = S * (1.0f / (float)H);
    float var  = Q * (1.0f / (float)H) - mean * mean;
    stat[0] = mean;
    stat[1] = rsqrtf(var + 1e-5f);
  }
  __syncthreads();
  const float mean = stat[0], rstd = stat[1];
  float4 gg = *(const float4*)(gamma + i0);
  float4 bb = *(const float4*)(beta + i0);
  _Float16* o = out + (size_t)r * H + i0;
  o[0] = (_Float16)((v.x - mean) * rstd * gg.x + bb.x);
  o[1] = (_Float16)((v.y - mean) * rstd * gg.y + bb.y);
  o[2] = (_Float16)((v.z - mean) * rstd * gg.z + bb.z);
  o[3] = (_Float16)((v.w - mean) * rstd * gg.w + bb.w);
}

// ---------------------------------------------------------------------------
// WMMA GEMM:  out[M,N] = A[M,K](f16) @ W[N,K]^T(f16) + bias1 (+bias2)
// Template: ACT 0 none / 1 relu / 2 x*sigmoid(1.702x); HASB2; WF/WH outputs.
// 256 threads = 8 wave32 waves; block tile 128x128; wave tile 64x32; BK=32.
// Staging: GLOBAL_LOAD_ASYNC_TO_LDS_B128 (ASYNCcnt) when available.
// Fragments use contiguous k = 16*(lane/16) + e for A and B (A 16x32, B 32x16, D 16x16 f32).
// ---------------------------------------------------------------------------
template <int ACT, bool HASB2, bool WF, bool WH>
__launch_bounds__(256, 2)
__global__ void gemm_wmma_kernel(const _Float16* __restrict__ A,
                                 const _Float16* __restrict__ W,
                                 const float* __restrict__ bias1,
                                 const float* __restrict__ bias2,
                                 float* __restrict__ outF,
                                 _Float16* __restrict__ outH,
                                 int M, int N, int K) {
  __shared__ __align__(16) _Float16 As[128 * 40];  // 32 cols + 8 pad (16B align kept)
  __shared__ __align__(16) _Float16 Bs[128 * 40];
  const int bm   = blockIdx.y * 128;
  const int bn   = blockIdx.x * 128;
  const int lane = threadIdx.x & 31;
  const int mi   = lane & 15;   // M (A) or N (B/D) index within tile
  const int hs   = lane >> 4;   // half-wave select
  const int wave = threadIdx.x >> 5;
  const int wm   = (wave >> 2) * 64;  // wave M offset (2 rows of waves)
  const int wn   = (wave & 3) * 32;   // wave N offset (4 cols of waves)
  const int tr   = threadIdx.x >> 1;          // staging row 0..127
  const int seg  = (threadIdx.x & 1) * 16;    // staging col seg 0/16 (halves)

  const _Float16* gA = A + (size_t)(bm + tr) * K + seg;
  const _Float16* gB = W + (size_t)(bn + tr) * K + seg;

  v8f acc[4][2] = {};

  for (int k0 = 0; k0 < K; k0 += 32) {
    __syncthreads();
#if USE_ASYNC_LDS
    {
      const uintptr_t ga = (uintptr_t)(gA + k0);
      const uintptr_t gb = (uintptr_t)(gB + k0);
      const uintptr_t la = (uintptr_t)(As + tr * 40 + seg);
      const uintptr_t lb = (uintptr_t)(Bs + tr * 40 + seg);
      __builtin_amdgcn_global_load_async_to_lds_b128((AS1 v4i*)ga,
                                                     (AS3 v4i*)la, 0, 0);
      __builtin_amdgcn_global_load_async_to_lds_b128((AS1 v4i*)(ga + 16),
                                                     (AS3 v4i*)(la + 16), 0, 0);
      __builtin_amdgcn_global_load_async_to_lds_b128((AS1 v4i*)gb,
                                                     (AS3 v4i*)lb, 0, 0);
      __builtin_amdgcn_global_load_async_to_lds_b128((AS1 v4i*)(gb + 16),
                                                     (AS3 v4i*)(lb + 16), 0, 0);
      __builtin_amdgcn_s_wait_asynccnt(0);
    }
#else
    {
      const _Float16* ga = gA + k0;
      *(v8h*)(As + tr * 40 + seg)     = *(const v8h*)(ga);
      *(v8h*)(As + tr * 40 + seg + 8) = *(const v8h*)(ga + 8);
      const _Float16* gb = gB + k0;
      *(v8h*)(Bs + tr * 40 + seg)     = *(const v8h*)(gb);
      *(v8h*)(Bs + tr * 40 + seg + 8) = *(const v8h*)(gb + 8);
    }
#endif
    if (k0 + 32 < K) {  // warm L2 for the next K tile (global_prefetch_b8)
      __builtin_prefetch(gA + k0 + 32, 0, 1);
      __builtin_prefetch(gB + k0 + 32, 0, 1);
    }
    __syncthreads();

    v16h bf[2];
#pragma unroll
    for (int fn = 0; fn < 2; ++fn) {
      const _Float16* bp = Bs + (wn + fn * 16 + mi) * 40 + 16 * hs;
      v8h lo = *(const v8h*)(bp);
      v8h hi = *(const v8h*)(bp + 8);
#pragma unroll
      for (int e = 0; e < 8; ++e) { bf[fn][e] = lo[e]; bf[fn][8 + e] = hi[e]; }
    }
#pragma unroll
    for (int fm = 0; fm < 4; ++fm) {
      const _Float16* ap = As + (wm + fm * 16 + mi) * 40;
      v8h lo = *(const v8h*)(ap + 16 * hs);
      v8h hi = *(const v8h*)(ap + 16 * hs + 8);
      v16h af;
#pragma unroll
      for (int e = 0; e < 8; ++e) { af[e] = lo[e]; af[8 + e] = hi[e]; }
#pragma unroll
      for (int fn = 0; fn < 2; ++fn)
        acc[fm][fn] = __builtin_amdgcn_wmma_f32_16x16x32_f16(
            false, af, false, bf[fn], (short)0, acc[fm][fn], false, false);
    }
  }

#pragma unroll
  for (int fm = 0; fm < 4; ++fm) {
#pragma unroll
    for (int fn = 0; fn < 2; ++fn) {
      const int col = bn + wn + fn * 16 + mi;
      float bsum = bias1[col];
      if (HASB2) bsum += bias2[col];
#pragma unroll
      for (int r = 0; r < 8; ++r) {
        const int row = bm + wm + fm * 16 + hs * 8 + r;
        float v = acc[fm][fn][r] + bsum;
        if (ACT == 1)      v = fmaxf(v, 0.0f);
        else if (ACT == 2) v = v * sigf(1.702f * v);
        const size_t off = (size_t)row * N + col;
        if (WF) outF[off] = v;
        if (WH) outH[off] = (_Float16)v;
      }
    }
  }
}

// ---------------------------------------------------------------------------
// Persistent LSTM recurrence: 128 sequential steps of g = h @ w_hh^T (+xg),
// gate nonlinearities, state update.  64 WGs x 256 threads; software grid
// barrier (monotonic counter, memset to 0 before launch).  Each wave does one
// 16x16 WMMA tile per k-step; w_hh (8.4 MB f16) stays L2-resident all steps.
// ---------------------------------------------------------------------------
__device__ __forceinline__ void grid_barrier(unsigned* cnt, unsigned goal) {
  __syncthreads();
  if (threadIdx.x == 0) {
    __threadfence();
    atomicAdd(cnt, 1u);
    while (atomicAdd(cnt, 0u) < goal) __builtin_amdgcn_s_sleep(1);
    __threadfence();
  }
  __syncthreads();
}

__launch_bounds__(256, 1)
__global__ void lstm_seq_kernel(const float* __restrict__ xg,       // [B,T,4H] (+b_ih+b_hh)
                                const _Float16* __restrict__ whh,   // [4H,H] f16
                                float* __restrict__ gbuf,           // [B,4H]
                                _Float16* __restrict__ hh,          // [B,H] f16 state
                                float* __restrict__ cst,            // [B,H] fp32 cell
                                float* __restrict__ hseqF,          // [B,T,H] fp32
                                _Float16* __restrict__ hseqH,       // [B,T,H] f16
                                unsigned* __restrict__ barrier_cnt) {
  const int gtid     = blockIdx.x * 256 + (int)threadIdx.x;
  const int nthreads = LSTM_BLOCKS * 256;
  for (int i = gtid; i < BB * H; i += nthreads) { hh[i] = (_Float16)0.0f; cst[i] = 0.0f; }
  unsigned bars = 1;
  grid_barrier(barrier_cnt, LSTM_BLOCKS * bars); bars++;

  const int lane  = threadIdx.x & 31;
  const int mi    = lane & 15;
  const int hs    = lane >> 4;
  const int wave  = threadIdx.x >> 5;
  const int m0    = (wave & 1) * 16;        // batch-row tile: 0 or 16
  const int nc    = (wave >> 1) * 16;       // 0,16,32,48 within WG's 64 cols
  const int nbase = blockIdx.x * 64;        // WG's slice of the 4H=4096 outputs

  for (int t = 0; t < T; ++t) {
    v8f acc = {};
    const _Float16* wrow = whh + (size_t)(nbase + nc + mi) * H;
    const _Float16* arow = hh + (size_t)(m0 + mi) * H;
    for (int k0 = 0; k0 < H; k0 += 32) {
      v8h alo = *(const v8h*)(arow + k0 + 16 * hs);
      v8h ahi = *(const v8h*)(arow + k0 + 16 * hs + 8);
      v8h blo = *(const v8h*)(wrow + k0 + 16 * hs);
      v8h bhi = *(const v8h*)(wrow + k0 + 16 * hs + 8);
      v16h af, bf;
#pragma unroll
      for (int e = 0; e < 8; ++e) {
        af[e] = alo[e]; af[8 + e] = ahi[e];
        bf[e] = blo[e]; bf[8 + e] = bhi[e];
      }
      acc = __builtin_amdgcn_wmma_f32_16x16x32_f16(false, af, false, bf,
                                                   (short)0, acc, false, false);
    }
#pragma unroll
    for (int r = 0; r < 8; ++r) {
      const int row = m0 + hs * 8 + r;
      gbuf[(size_t)row * (4 * H) + nbase + nc + mi] = acc[r];
    }
    grid_barrier(barrier_cnt, LSTM_BLOCKS * bars); bars++;

    for (int i = gtid; i < BB * H; i += nthreads) {
      const int b = i >> 10;
      const int j = i & (H - 1);
      const size_t gb = (size_t)b * (4 * H);
      const size_t xb = (size_t)(b * T + t) * (4 * H);
      float gi = gbuf[gb + j]           + xg[xb + j];
      float gf = gbuf[gb + H + j]       + xg[xb + H + j];
      float gg = gbuf[gb + 2 * H + j]   + xg[xb + 2 * H + j];
      float go = gbuf[gb + 3 * H + j]   + xg[xb + 3 * H + j];
      float c = sigf(gf) * cst[i] + sigf(gi) * tanhf(gg);
      float h = sigf(go) * tanhf(c);
      cst[i] = c;
      hh[i]  = (_Float16)h;
      const size_t so = (size_t)(b * T + t) * H + j;
      hseqF[so] = h;
      hseqH[so] = (_Float16)h;
    }
    grid_barrier(barrier_cnt, LSTM_BLOCKS * bars); bars++;
  }
}

// ---------------------------------------------------------------------------
// kbuf[B,148,H] = concat(zeros[B,20,H], relu(hseq))
// ---------------------------------------------------------------------------
__global__ void pad_relu_kernel(const float* __restrict__ hseq,
                                float* __restrict__ kbuf) {
  int i = blockIdx.x * 256 + threadIdx.x;
  const int total = MK * H / 4;
  if (i >= total) return;
  const int idx4 = i * 4;
  const int row = idx4 >> 10;
  const int col = idx4 & (H - 1);
  const int b = row / KROWS, tt = row % KROWS;
  float4 v;
  if (tt < CACHE) {
    v = make_float4(0.f, 0.f, 0.f, 0.f);
  } else {
    v = *(const float4*)(hseq + (size_t)(b * T + tt - CACHE) * H + col);
    v.x = fmaxf(v.x, 0.f); v.y = fmaxf(v.y, 0.f);
    v.z = fmaxf(v.z, 0.f); v.w = fmaxf(v.w, 0.f);
  }
  *(float4*)(kbuf + (size_t)idx4) = v;
}

// kv = relu(kp) * vp  (in place into kp)
__global__ void kv_mul_kernel(float* __restrict__ kp,
                              const float* __restrict__ vp, int n4) {
  int i = blockIdx.x * 256 + threadIdx.x;
  if (i >= n4) return;
  float4 a = *(const float4*)(kp + (size_t)i * 4);
  float4 b = *(const float4*)(vp + (size_t)i * 4);
  a.x = fmaxf(a.x, 0.f) * b.x; a.y = fmaxf(a.y, 0.f) * b.y;
  a.z = fmaxf(a.z, 0.f) * b.z; a.w = fmaxf(a.w, 0.f) * b.w;
  *(float4*)(kp + (size_t)i * 4) = a;
}

// out_f16[b,qi,d] = relu(qp) * sum_{k=qi+1..qi+20} kv[b,k,d]
__global__ void band_kernel(const float* __restrict__ kv,
                            const float* __restrict__ qp,
                            _Float16* __restrict__ outH) {
  int i = blockIdx.x * 256 + threadIdx.x;
  if (i >= MQ * H) return;
  const int row = i >> 10, d = i & (H - 1);
  const int b = row >> 7, qi = row & (T - 1);
  const float* base = kv + (size_t)(b * KROWS + qi + 1) * H + d;
  float w = 0.f;
#pragma unroll
  for (int k = 0; k < CACHE; ++k) w += base[(size_t)k * H];
  const float qv = fmaxf(qp[(size_t)row * H + d], 0.0f);
  outH[(size_t)i] = (_Float16)(qv * w);
}

// ---------------------------------------------------------------------------
// Orchestration
// ---------------------------------------------------------------------------
extern "C" void kernel_launch(void* const* d_in, const int* in_sizes, int n_in,
                              void* d_out, int out_size, void* d_ws, size_t ws_size,
                              hipStream_t stream) {
  (void)in_sizes; (void)n_in; (void)out_size; (void)ws_size;
  const float* vec = (const float*)d_in[0];

  char* base = (char*)d_ws;
  size_t off = 0;
  auto alloc = [&](size_t bytes) -> void* {
    void* p = base + off;
    off += (bytes + 255) & ~(size_t)255;
    return p;
  };

  _Float16* wih0_16 = (_Float16*)alloc((size_t)4 * H * H * 2);
  _Float16* whh0_16 = (_Float16*)alloc((size_t)4 * H * H * 2);
  _Float16* wih1_16 = (_Float16*)alloc((size_t)4 * H * H * 2);
  _Float16* whh1_16 = (_Float16*)alloc((size_t)4 * H * H * 2);
  _Float16* wq16    = (_Float16*)alloc((size_t)H * H * 2);
  _Float16* wk16    = (_Float16*)alloc((size_t)H * H * 2);
  _Float16* wv16    = (_Float16*)alloc((size_t)H * H * 2);
  _Float16* w1_16   = (_Float16*)alloc((size_t)2 * H * H * 2);
  _Float16* w2_16   = (_Float16*)alloc((size_t)2 * H * H * 2);
  float*    xg     = (float*)   alloc((size_t)MQ * 4 * H * 4);
  _Float16* act16  = (_Float16*)alloc((size_t)MK * H * 2);
  _Float16* h1a16  = (_Float16*)alloc((size_t)MQ * 2 * H * 2);
  float*    hseqF  = (float*)   alloc((size_t)MQ * H * 4);
  _Float16* hseqH  = (_Float16*)alloc((size_t)MQ * H * 2);
  float*    kbuf   = (float*)   alloc((size_t)MK * H * 4);
  float*    qp     = (float*)   alloc((size_t)MQ * H * 4);
  float*    kp     = (float*)   alloc((size_t)MK * H * 4);
  float*    vp     = (float*)   alloc((size_t)MK * H * 4);
  float*    qcur   = (float*)   alloc((size_t)MQ * H * 4);
  float*    gbuf   = (float*)   alloc((size_t)BB * 4 * H * 4);
  _Float16* hh16   = (_Float16*)alloc((size_t)BB * H * 2);
  float*    cst    = (float*)   alloc((size_t)BB * H * 4);
  unsigned* bar    = (unsigned*)alloc(256);

  auto conv = [&](const float* src, _Float16* dst, size_t n) {
    int n4 = (int)(n / 4);
    f32_to_f16_kernel<<<(n4 + 255) / 256, 256, 0, stream>>>(src, dst, n4);
  };

  const float* qin = vec;
  for (int layer = 0; layer < 2; ++layer) {
    const int L = 1 + layer * 26;
    const float* rnln_g = (const float*)d_in[L + 0];
    const float* rnln_b = (const float*)d_in[L + 1];
    const float* w_ih0  = (const float*)d_in[L + 2];
    const float* w_hh0  = (const float*)d_in[L + 3];
    const float* b_ih0  = (const float*)d_in[L + 4];
    const float* b_hh0  = (const float*)d_in[L + 5];
    const float* w_ih1  = (const float*)d_in[L + 6];
    const float* w_hh1  = (const float*)d_in[L + 7];
    const float* b_ih1  = (const float*)d_in[L + 8];
    const float* b_hh1  = (const float*)d_in[L + 9];
    const float* lnq_g  = (const float*)d_in[L + 10];
    const float* lnq_b  = (const float*)d_in[L + 11];
    const float* lnk_g  = (const float*)d_in[L + 12];
    const float* lnk_b  = (const float*)d_in[L + 13];
    const float* lnv_g  = (const float*)d_in[L + 14];
    const float* lnv_b  = (const float*)d_in[L + 15];
    const float* wq_w   = (const float*)d_in[L + 16];
    const float* wq_b   = (const float*)d_in[L + 17];
    const float* wk_w   = (const float*)d_in[L + 18];
    const float* wk_b   = (const float*)d_in[L + 19];
    const float* wv_w   = (const float*)d_in[L + 20];
    const float* wv_b   = (const float*)d_in[L + 21];
    const float* bw1    = (const float*)d_in[L + 22];
    const float* bb1    = (const float*)d_in[L + 23];
    const float* bw2    = (const float*)d_in[L + 24];
    const float* bb2    = (const float*)d_in[L + 25];

    conv(w_ih0, wih0_16, (size_t)4 * H * H);
    conv(w_hh0, whh0_16, (size_t)4 * H * H);
    conv(w_ih1, wih1_16, (size_t)4 * H * H);
    conv(w_hh1, whh1_16, (size_t)4 * H * H);
    conv(wq_w, wq16, (size_t)H * H);
    conv(wk_w, wk16, (size_t)H * H);
    conv(wv_w, wv16, (size_t)H * H);
    conv(bw1, w1_16, (size_t)2 * H * H);
    conv(bw2, w2_16, (size_t)2 * H * H);

    // rnln -> f16
    layernorm_kernel<<<MQ, 256, 0, stream>>>(qin, rnln_g, rnln_b, act16, 0);

    // LSTM sub 0: big input-projection GEMM, then persistent recurrence
    dim3 g_xg(4 * H / 128, MQ / 128);
    gemm_wmma_kernel<0, true, true, false><<<g_xg, 256, 0, stream>>>(
        act16, wih0_16, b_ih0, b_hh0, xg, nullptr, MQ, 4 * H, H);
    (void)hipMemsetAsync(bar, 0, sizeof(unsigned), stream);
    lstm_seq_kernel<<<LSTM_BLOCKS, 256, 0, stream>>>(xg, whh0_16, gbuf, hh16, cst,
                                                     hseqF, hseqH, bar);
    // LSTM sub 1
    gemm_wmma_kernel<0, true, true, false><<<g_xg, 256, 0, stream>>>(
        hseqH, wih1_16, b_ih1, b_hh1, xg, nullptr, MQ, 4 * H, H);
    (void)hipMemsetAsync(bar, 0, sizeof(unsigned), stream);
    lstm_seq_kernel<<<LSTM_BLOCKS, 256, 0, stream>>>(xg, whh1_16, gbuf, hh16, cst,
                                                     hseqF, hseqH, bar);

    // relu + left zero-pad to K=148
    pad_relu_kernel<<<(MK * H / 4 + 255) / 256, 256, 0, stream>>>(hseqF, kbuf);

    // projections (LN -> f16 -> WMMA GEMM), reusing act16 sequentially
    dim3 g_q(H / 128, MQ / 128), g_k(H / 128, MK / 128);
    layernorm_kernel<<<MQ, 256, 0, stream>>>(kbuf, lnq_g, lnq_b, act16, 1);
    gemm_wmma_kernel<0, false, true, false><<<g_q, 256, 0, stream>>>(
        act16, wq16, wq_b, nullptr, qp, nullptr, MQ, H, H);
    layernorm_kernel<<<MK, 256, 0, stream>>>(kbuf, lnk_g, lnk_b, act16, 0);
    gemm_wmma_kernel<0, false, true, false><<<g_k, 256, 0, stream>>>(
        act16, wk16, wk_b, nullptr, kp, nullptr, MK, H, H);
    layernorm_kernel<<<MK, 256, 0, stream>>>(kbuf, lnv_g, lnv_b, act16, 0);
    gemm_wmma_kernel<0, false, true, false><<<g_k, 256, 0, stream>>>(
        act16, wv16, wv_b, nullptr, vp, nullptr, MK, H, H);

    // kv = relu(kp)*vp (in place), then banded window + relu(qp)* -> f16
    kv_mul_kernel<<<(MK * H / 4 + 255) / 256, 256, 0, stream>>>(kp, vp, MK * H / 4);
    band_kernel<<<(MQ * H + 255) / 256, 256, 0, stream>>>(kp, qp, act16);

    // Boom MLP: H->2H with x*sigmoid(1.702x) (f16 out), then 2H->H (fp32 out)
    dim3 g_b1(2 * H / 128, MQ / 128), g_b2(H / 128, MQ / 128);
    gemm_wmma_kernel<2, false, false, true><<<g_b1, 256, 0, stream>>>(
        act16, w1_16, bb1, nullptr, nullptr, h1a16, MQ, 2 * H, H);
    float* outp = (layer == 1) ? (float*)d_out : qcur;
    gemm_wmma_kernel<0, false, true, false><<<g_b2, 256, 0, stream>>>(
        h1a16, w2_16, bb2, nullptr, outp, nullptr, MQ, H, 2 * H);
    qin = qcur;
  }
}
